// FT1d_43671227465838
// MI455X (gfx1250) — matrix operand order repaired and checked by reading, not compiled, not run on hardware
//
#include <hip/hip_runtime.h>
#include <hip/hip_bf16.h>

typedef __attribute__((ext_vector_type(16))) _Float16 v16h;
typedef __attribute__((ext_vector_type(8)))  _Float16 v8h;
typedef __attribute__((ext_vector_type(8)))  float    v8f;

#define T_ALL   131072   // B*N = 16*8192
#define T_SUB   16384    // B*1024
#define NSEQ    8192
#define NSUB    1024

static __device__ __forceinline__ float wave_sum(float v) {
    #pragma unroll
    for (int off = 16; off > 0; off >>= 1) v += __shfl_xor(v, off, 32);
    return v;
}

// gfx1250 async global->LDS copy (16B per lane), tracked by ASYNCcnt
static __device__ __forceinline__ void async_copy16(unsigned lds_addr, const void* g) {
    asm volatile("global_load_async_to_lds_b128 %0, %1, off"
                 :: "v"(lds_addr), "v"((unsigned long long)(size_t)g)
                 : "memory");
}
static __device__ __forceinline__ void wait_async0() {
    asm volatile("s_wait_asynccnt 0x0" ::: "memory");
}

// ---------------------------------------------------------------------------
// Weight conversion: src (Kd x Nd) f32 row-major  ->  dst (Nd x Kd) f16
// ---------------------------------------------------------------------------
__global__ void k_wcvt(const float* __restrict__ src, _Float16* __restrict__ dst,
                       int Kd, int Nd) {
    int i = blockIdx.x * 256 + threadIdx.x;
    if (i >= Kd * Nd) return;
    int n = i / Kd, k = i % Kd;
    dst[i] = (_Float16)src[(size_t)k * Nd + n];
}

// conv_w (O=128, I=128, S=3) f32 -> dst[o][s*128+i] f16 (im2col B layout)
__global__ void k_convcvt(const float* __restrict__ src, _Float16* __restrict__ dst) {
    int i = blockIdx.x * 256 + threadIdx.x;
    if (i >= 128 * 384) return;
    int o = i / 384, rem = i % 384, s = rem / 128, ic = rem % 128;
    dst[i] = (_Float16)src[((size_t)o * 128 + ic) * 3 + s];
}

// ---------------------------------------------------------------------------
// Embed: h[t][c] = a[t]*p_w[0][c] + x[t]*p_w[1][c] + p_b[c]   (f16 out)
// ---------------------------------------------------------------------------
__global__ void k_embed(const float* __restrict__ x, const float* __restrict__ a,
                        const float* __restrict__ pw, const float* __restrict__ pb,
                        _Float16* __restrict__ H) {
    int i = blockIdx.x * 256 + threadIdx.x;       // over T_ALL*128
    int t = i >> 7, c = i & 127;
    H[i] = (_Float16)(a[t] * pw[c] + x[t] * pw[128 + c] + pb[c]);
}

// ---------------------------------------------------------------------------
// Generic token GEMM: Y[M x NC] = act(X[M x 128] @ W + bias (+res)), f16 I/O.
// Wt is f16 in [NC][128] layout (transposed).  M must be a multiple of 64.
// X tile staged via GLOBAL_LOAD_ASYNC_TO_LDS_B128.
// ---------------------------------------------------------------------------
template<int RELU, int HAS_RES>
__global__ __launch_bounds__(256)
void k_gemm128(const _Float16* __restrict__ X, const _Float16* __restrict__ Wt,
               const float* __restrict__ bias, const _Float16* __restrict__ res,
               _Float16* __restrict__ Y, int M, int NC) {
    constexpr int K = 128, XSTR = K + 8;          // 272B row stride in LDS
    __shared__ _Float16 Xs[64 * XSTR];
    const int m0 = blockIdx.x * 64;

    // async-stage 64 x 128 f16 tile (16B chunks, padded stride)
    const unsigned lds_base = (unsigned)(size_t)&Xs[0];
    #pragma unroll
    for (int it = 0; it < 4; ++it) {
        int q = threadIdx.x + 256 * it;           // 1024 chunks total
        int r = q >> 4, c16 = q & 15;
        unsigned lds = lds_base + r * (XSTR * 2) + c16 * 16;
        async_copy16(lds, X + (size_t)(m0 + r) * K + c16 * 8);
    }
    wait_async0();
    __syncthreads();

    const int wave = threadIdx.x >> 5, lane = threadIdx.x & 31;
    const int g = lane >> 4, l16 = lane & 15;

    for (int nt = wave; nt < (NC >> 4); nt += 8) {
        const int n0 = nt << 4;
        const _Float16* wrow = Wt + (size_t)(n0 + l16) * K;
        __builtin_prefetch(wrow, 0, 0);
        const float bv = bias[n0 + l16];
        for (int ms = 0; ms < 4; ++ms) {
            v8f c = {};
            const _Float16* arow = &Xs[(ms * 16 + l16) * XSTR];
            #pragma unroll
            for (int kc = 0; kc < 4; ++kc) {
                const int kk = kc * 32;
                v16h av, bvv;
                ((v8h*)&av)[0]  = *(const v8h*)(arow + kk + 16 * g);
                ((v8h*)&av)[1]  = *(const v8h*)(arow + kk + 16 * g + 8);
                ((v8h*)&bvv)[0] = *(const v8h*)(wrow + kk + 16 * g);
                ((v8h*)&bvv)[1] = *(const v8h*)(wrow + kk + 16 * g + 8);
                c = __builtin_amdgcn_wmma_f32_16x16x32_f16(
                        false, av, false, bvv, (short)0, c, false, false);
            }
            #pragma unroll
            for (int r = 0; r < 8; ++r) {
                int row = m0 + ms * 16 + r + 8 * g;
                if (row >= M) continue;
                size_t o = (size_t)row * NC + n0 + l16;
                float v = c[r] + bv;
                if (HAS_RES) v += (float)res[o];
                if (RELU) v = fmaxf(v, 0.0f);
                Y[o] = (_Float16)v;
            }
        }
    }
}

// ---------------------------------------------------------------------------
// Conv1d (k=3, pad=1) as im2col GEMM: K=384.  Wt layout [o][s*128+i], f16 I/O.
// Body rows async-staged; halo (zero-pad) rows DS-stored as zeros.
// ---------------------------------------------------------------------------
__global__ __launch_bounds__(256)
void k_conv_gemm(const _Float16* __restrict__ H, const _Float16* __restrict__ Wt,
                 const float* __restrict__ bias, _Float16* __restrict__ Y) {
    constexpr int K = 384, XSTR = K + 8;          // 784B row stride in LDS
    __shared__ _Float16 Xs[64 * XSTR];
    const int m0 = blockIdx.x * 64;
    const int b = m0 / NSEQ;
    const int nbase = m0 % NSEQ;

    const unsigned lds_base = (unsigned)(size_t)&Xs[0];
    #pragma unroll
    for (int it = 0; it < 12; ++it) {
        int q = threadIdx.x + 256 * it;           // 64 rows * 3 segs * 16 chunks
        int r = q / 48, rem = q % 48;
        int s = rem >> 4, c16 = rem & 15;
        int n = nbase + r + (s - 1);
        if (n >= 0 && n < NSEQ) {
            unsigned lds = lds_base + r * (XSTR * 2) + s * 256 + c16 * 16;
            async_copy16(lds, H + ((size_t)b * NSEQ + n) * 128 + c16 * 8);
        } else {
            v8h z = {};
            *(v8h*)&Xs[r * XSTR + s * 128 + c16 * 8] = z;
        }
    }
    wait_async0();
    __syncthreads();

    const int wave = threadIdx.x >> 5, lane = threadIdx.x & 31;
    const int g = lane >> 4, l16 = lane & 15;
    const int n0 = wave << 4;
    const _Float16* wrow = Wt + (size_t)(n0 + l16) * K;
    __builtin_prefetch(wrow, 0, 0);
    const float bv = bias[n0 + l16];

    for (int ms = 0; ms < 4; ++ms) {
        v8f c = {};
        const _Float16* arow = &Xs[(ms * 16 + l16) * XSTR];
        #pragma unroll
        for (int kc = 0; kc < 12; ++kc) {
            const int kk = kc * 32;
            v16h av, bvv;
            ((v8h*)&av)[0]  = *(const v8h*)(arow + kk + 16 * g);
            ((v8h*)&av)[1]  = *(const v8h*)(arow + kk + 16 * g + 8);
            ((v8h*)&bvv)[0] = *(const v8h*)(wrow + kk + 16 * g);
            ((v8h*)&bvv)[1] = *(const v8h*)(wrow + kk + 16 * g + 8);
            c = __builtin_amdgcn_wmma_f32_16x16x32_f16(
                    false, av, false, bvv, (short)0, c, false, false);
        }
        #pragma unroll
        for (int r = 0; r < 8; ++r) {
            int row = m0 + ms * 16 + r + 8 * g;
            Y[(size_t)row * 128 + n0 + l16] = (_Float16)(c[r] + bv);
        }
    }
}

// ---------------------------------------------------------------------------
// Subsample h[:, ::8] -> xc (f16)
// ---------------------------------------------------------------------------
__global__ void k_subsample(const _Float16* __restrict__ H, _Float16* __restrict__ Xc) {
    int i = blockIdx.x * 256 + threadIdx.x;       // over T_SUB*128
    int t = i >> 7, c = i & 127;
    int b = t >> 10, j = t & 1023;
    Xc[i] = H[((size_t)b * NSEQ + j * 8) * 128 + c];
}

// ---------------------------------------------------------------------------
// Per-head LayerNorm over DH=32 (one wave per (token, head)), f16 I/O
// ---------------------------------------------------------------------------
__global__ __launch_bounds__(128)
void k_headln(_Float16* __restrict__ Q, const float* __restrict__ gw,
              const float* __restrict__ bw) {
    int wid = blockIdx.x * 4 + (threadIdx.x >> 5);
    int lane = threadIdx.x & 31;
    int t = wid >> 2, h = wid & 3;
    size_t o = (size_t)t * 128 + h * 32 + lane;
    float v = (float)Q[o];
    float m = wave_sum(v) * (1.0f / 32.0f);
    float d = v - m;
    float var = wave_sum(d * d) * (1.0f / 32.0f);
    Q[o] = (_Float16)(d * rsqrtf(var + 1e-5f) * gw[h * 32 + lane] + bw[h * 32 + lane]);
}

// ---------------------------------------------------------------------------
// kv[b,h] = (k^T @ v)/1024, output f16 as kvT[bh][dj][di] (B-fragment layout)
// ---------------------------------------------------------------------------
__global__ __launch_bounds__(128)
void k_kv(const _Float16* __restrict__ Kb, const _Float16* __restrict__ Vb,
          _Float16* __restrict__ kvT) {
    __shared__ _Float16 kT[32 * 40];
    __shared__ _Float16 vT[32 * 40];
    const int bh = blockIdx.x, b = bh >> 2, h = bh & 3;
    const int wave = threadIdx.x >> 5, lane = threadIdx.x & 31;
    const int g = lane >> 4, l16 = lane & 15;
    const int m0 = (wave >> 1) * 16;   // di tile
    const int n0 = (wave & 1) * 16;    // dj tile
    v8f c = {};
    for (int j0 = 0; j0 < NSUB; j0 += 32) {
        for (int idx = threadIdx.x; idx < 1024; idx += 128) {
            int tok = idx >> 5, d = idx & 31;
            size_t o = ((size_t)(b * NSUB + j0 + tok)) * 128 + h * 32 + d;
            kT[d * 40 + tok] = Kb[o];
            vT[d * 40 + tok] = Vb[o];
        }
        __syncthreads();
        v16h av, bvv;
        ((v8h*)&av)[0]  = *(const v8h*)(&kT[(m0 + l16) * 40 + 16 * g]);
        ((v8h*)&av)[1]  = *(const v8h*)(&kT[(m0 + l16) * 40 + 16 * g + 8]);
        ((v8h*)&bvv)[0] = *(const v8h*)(&vT[(n0 + l16) * 40 + 16 * g]);
        ((v8h*)&bvv)[1] = *(const v8h*)(&vT[(n0 + l16) * 40 + 16 * g + 8]);
        c = __builtin_amdgcn_wmma_f32_16x16x32_f16(
                false, av, false, bvv, (short)0, c, false, false);
        __syncthreads();
    }
    const float sc = 1.0f / 1024.0f;
    #pragma unroll
    for (int r = 0; r < 8; ++r) {
        int di = m0 + r + 8 * g, dj = n0 + l16;
        kvT[((size_t)bh * 32 + dj) * 32 + di] = (_Float16)(c[r] * sc);
    }
}

// ---------------------------------------------------------------------------
// attn[t, h*32+dj] = q[t, h*32+:] @ kv[b,h]   (K=32, single WMMA per tile)
// ---------------------------------------------------------------------------
__global__ __launch_bounds__(256)
void k_attn_ov(const _Float16* __restrict__ Q, const _Float16* __restrict__ kvT,
               _Float16* __restrict__ A) {
    const int m0 = blockIdx.x * 64;
    const int b = m0 >> 10;
    const int wave = threadIdx.x >> 5, lane = threadIdx.x & 31;
    const int g = lane >> 4, l16 = lane & 15;
    const int h = wave >> 1;
    const int n0 = (wave & 1) * 16;
    const _Float16* kvrow = kvT + (((size_t)(b * 4 + h) * 32) + n0 + l16) * 32 + 16 * g;
    v16h bvv;
    ((v8h*)&bvv)[0] = *(const v8h*)(kvrow);
    ((v8h*)&bvv)[1] = *(const v8h*)(kvrow + 8);
    for (int ms = 0; ms < 4; ++ms) {
        int row = m0 + ms * 16 + l16;
        const _Float16* qp = Q + (size_t)row * 128 + h * 32;
        v16h av;
        ((v8h*)&av)[0] = *(const v8h*)(qp + 16 * g);
        ((v8h*)&av)[1] = *(const v8h*)(qp + 16 * g + 8);
        v8f c = {};
        c = __builtin_amdgcn_wmma_f32_16x16x32_f16(
                false, av, false, bvv, (short)0, c, false, false);
        #pragma unroll
        for (int r = 0; r < 8; ++r) {
            int rr = m0 + ms * 16 + r + 8 * g;
            A[(size_t)rr * 128 + h * 32 + n0 + l16] = (_Float16)c[r];
        }
    }
}

// ---------------------------------------------------------------------------
// h = LN( x1 + interp_linear(x2, N) )  — one wave per output token, f16 I/O
// ---------------------------------------------------------------------------
__global__ __launch_bounds__(256)
void k_interp_add_ln(const _Float16* __restrict__ X1, const _Float16* __restrict__ X2,
                     const float* __restrict__ gw, const float* __restrict__ bw,
                     _Float16* __restrict__ Hout) {
    int wid = blockIdx.x * 8 + (threadIdx.x >> 5);
    int lane = threadIdx.x & 31;
    int b = wid >> 13, n = wid & (NSEQ - 1);
    float pos = (n + 0.5f) * 0.125f - 0.5f;
    pos = fminf(fmaxf(pos, 0.0f), 1023.0f);
    int i0 = (int)floorf(pos);
    int i1 = min(i0 + 1, 1023);
    float w = pos - (float)i0;
    const _Float16* r0 = X2 + ((size_t)b * NSUB + i0) * 128;
    const _Float16* r1 = X2 + ((size_t)b * NSUB + i1) * 128;
    const _Float16* rx = X1 + (size_t)wid * 128;
    float v[4]; float s = 0.0f;
    #pragma unroll
    for (int i = 0; i < 4; ++i) {
        int c = i * 32 + lane;
        v[i] = (float)rx[c] + (float)r0[c] * (1.0f - w) + (float)r1[c] * w;
        s += v[i];
    }
    float m = wave_sum(s) * (1.0f / 128.0f);
    float ss = 0.0f;
    #pragma unroll
    for (int i = 0; i < 4; ++i) { v[i] -= m; ss += v[i] * v[i]; }
    float inv = rsqrtf(wave_sum(ss) * (1.0f / 128.0f) + 1e-5f);
    #pragma unroll
    for (int i = 0; i < 4; ++i) {
        int c = i * 32 + lane;
        Hout[(size_t)wid * 128 + c] = (_Float16)(v[i] * inv * gw[c] + bw[c]);
    }
}

// ---------------------------------------------------------------------------
// Y = LN(X) over 128 — one wave per token, f16 I/O
// ---------------------------------------------------------------------------
__global__ __launch_bounds__(256)
void k_ln(const _Float16* __restrict__ X, const float* __restrict__ gw,
          const float* __restrict__ bw, _Float16* __restrict__ Y) {
    int wid = blockIdx.x * 8 + (threadIdx.x >> 5);
    int lane = threadIdx.x & 31;
    const _Float16* r = X + (size_t)wid * 128;
    float v[4]; float s = 0.0f;
    #pragma unroll
    for (int i = 0; i < 4; ++i) { v[i] = (float)r[i * 32 + lane]; s += v[i]; }
    float m = wave_sum(s) * (1.0f / 128.0f);
    float ss = 0.0f;
    #pragma unroll
    for (int i = 0; i < 4; ++i) { v[i] -= m; ss += v[i] * v[i]; }
    float inv = rsqrtf(wave_sum(ss) * (1.0f / 128.0f) + 1e-5f);
    #pragma unroll
    for (int i = 0; i < 4; ++i) {
        int c = i * 32 + lane;
        Y[(size_t)wid * 128 + c] = (_Float16)(v[i] * inv * gw[c] + bw[c]);
    }
}

// ---------------------------------------------------------------------------
// Final head: out[t] = relu_tmp(f16, 256) . q_w2 + q_b2 — one wave per token
// ---------------------------------------------------------------------------
__global__ __launch_bounds__(256)
void k_final(const _Float16* __restrict__ Tmp, const float* __restrict__ w2,
             const float* __restrict__ b2, float* __restrict__ out) {
    int wid = blockIdx.x * 8 + (threadIdx.x >> 5);
    int lane = threadIdx.x & 31;
    const _Float16* r = Tmp + (size_t)wid * 256;
    float s = 0.0f;
    #pragma unroll
    for (int i = 0; i < 8; ++i) {
        int c = i * 32 + lane;
        s += (float)r[c] * w2[c];
    }
    s = wave_sum(s);
    if (lane == 0) out[wid] = s + b2[0];
}

// ===========================================================================
extern "C" void kernel_launch(void* const* d_in, const int* in_sizes, int n_in,
                              void* d_out, int out_size, void* d_ws, size_t ws_size,
                              hipStream_t stream) {
    (void)in_sizes; (void)n_in; (void)out_size; (void)ws_size;
    const float* x     = (const float*)d_in[0];
    const float* a     = (const float*)d_in[1];
    const float* p_w   = (const float*)d_in[2];
    const float* p_b   = (const float*)d_in[3];
    const float* wq    = (const float*)d_in[4];
    const float* bq    = (const float*)d_in[5];
    const float* wk    = (const float*)d_in[6];
    const float* bk    = (const float*)d_in[7];
    const float* wv    = (const float*)d_in[8];
    const float* bv    = (const float*)d_in[9];
    const float* wo    = (const float*)d_in[10];
    const float* bo    = (const float*)d_in[11];
    const float* lnq_g = (const float*)d_in[12];
    const float* lnq_b = (const float*)d_in[13];
    const float* lnk_g = (const float*)d_in[14];
    const float* lnk_b = (const float*)d_in[15];
    const float* convw = (const float*)d_in[16];
    const float* convb = (const float*)d_in[17];
    const float* ln1_g = (const float*)d_in[18];
    const float* ln1_b = (const float*)d_in[19];
    const float* ln2_g = (const float*)d_in[20];
    const float* ln2_b = (const float*)d_in[21];
    const float* fw1   = (const float*)d_in[22];
    const float* fb1   = (const float*)d_in[23];
    const float* fw2   = (const float*)d_in[24];
    const float* fb2   = (const float*)d_in[25];
    const float* qw1   = (const float*)d_in[26];
    const float* qb1   = (const float*)d_in[27];
    const float* qw2   = (const float*)d_in[28];
    const float* qb2   = (const float*)d_in[29];

    char* p = (char*)d_ws;
    auto carve = [&](size_t bytes) -> char* {
        char* r = p; p += (bytes + 255) & ~(size_t)255; return r;
    };
    _Float16* Wq16  = (_Float16*)carve(4 * 128 * 128 * 2);
    _Float16* Wk16  = (_Float16*)carve(4 * 128 * 128 * 2);
    _Float16* Wv16  = (_Float16*)carve(4 * 128 * 128 * 2);
    _Float16* Wo16  = (_Float16*)carve(4 * 128 * 128 * 2);
    _Float16* Wf1   = (_Float16*)carve(4 * 128 * 128 * 2);
    _Float16* Wf2   = (_Float16*)carve(4 * 128 * 128 * 2);
    _Float16* Wc16  = (_Float16*)carve(4 * 128 * 384 * 2);
    _Float16* Wqw1  = (_Float16*)carve(256 * 128 * 2);
    _Float16* kvT   = (_Float16*)carve(64 * 32 * 32 * 2);
    _Float16* bh    = (_Float16*)carve((size_t)T_ALL * 128 * 2);
    _Float16* bx1   = (_Float16*)carve((size_t)T_ALL * 128 * 2);
    _Float16* bt    = (_Float16*)carve((size_t)T_ALL * 256 * 2);  // FFN tmp / head tmp
    _Float16* bxc   = (_Float16*)carve((size_t)T_SUB * 128 * 2);
    _Float16* bq_   = (_Float16*)carve((size_t)T_SUB * 128 * 2);
    _Float16* bk_   = (_Float16*)carve((size_t)T_SUB * 128 * 2);
    _Float16* bv_   = (_Float16*)carve((size_t)T_SUB * 128 * 2);
    _Float16* batn  = (_Float16*)carve((size_t)T_SUB * 128 * 2);
    _Float16* bx2   = (_Float16*)carve((size_t)T_SUB * 128 * 2);

    const int G1 = (128 * 128 + 255) / 256;
    for (int i = 0; i < 4; ++i) {
        k_wcvt<<<G1, 256, 0, stream>>>(wq + i * 16384, Wq16 + i * 16384, 128, 128);
        k_wcvt<<<G1, 256, 0, stream>>>(wk + i * 16384, Wk16 + i * 16384, 128, 128);
        k_wcvt<<<G1, 256, 0, stream>>>(wv + i * 16384, Wv16 + i * 16384, 128, 128);
        k_wcvt<<<G1, 256, 0, stream>>>(wo + i * 16384, Wo16 + i * 16384, 128, 128);
        k_wcvt<<<G1, 256, 0, stream>>>(fw1 + i * 16384, Wf1 + i * 16384, 128, 128);
        k_wcvt<<<G1, 256, 0, stream>>>(fw2 + i * 16384, Wf2 + i * 16384, 128, 128);
        k_convcvt<<<(128 * 384 + 255) / 256, 256, 0, stream>>>(
            convw + (size_t)i * 128 * 128 * 3, Wc16 + i * 49152);
    }
    k_wcvt<<<(128 * 256 + 255) / 256, 256, 0, stream>>>(qw1, Wqw1, 128, 256);

    k_embed<<<(T_ALL * 128) / 256, 256, 0, stream>>>(x, a, p_w, p_b, bh);

    for (int i = 0; i < 4; ++i) {
        k_conv_gemm<<<T_ALL / 64, 256, 0, stream>>>(bh, Wc16 + i * 49152,
                                                    convb + i * 128, bx1);
        k_subsample<<<(T_SUB * 128) / 256, 256, 0, stream>>>(bh, bxc);
        k_gemm128<0,0><<<T_SUB / 64, 256, 0, stream>>>(
            bxc, Wq16 + i * 16384, bq + i * 128, nullptr, bq_, T_SUB, 128);
        k_gemm128<0,0><<<T_SUB / 64, 256, 0, stream>>>(
            bxc, Wk16 + i * 16384, bk + i * 128, nullptr, bk_, T_SUB, 128);
        k_gemm128<0,0><<<T_SUB / 64, 256, 0, stream>>>(
            bxc, Wv16 + i * 16384, bv + i * 128, nullptr, bv_, T_SUB, 128);
        k_headln<<<(T_SUB * 4) / 4, 128, 0, stream>>>(bq_, lnq_g + i * 128, lnq_b + i * 128);
        k_headln<<<(T_SUB * 4) / 4, 128, 0, stream>>>(bk_, lnk_g + i * 128, lnk_b + i * 128);
        k_kv<<<64, 128, 0, stream>>>(bk_, bv_, kvT);
        k_attn_ov<<<T_SUB / 64, 256, 0, stream>>>(bq_, kvT, batn);
        k_gemm128<0,0><<<T_SUB / 64, 256, 0, stream>>>(
            batn, Wo16 + i * 16384, bo + i * 128, nullptr, bx2, T_SUB, 128);
        k_interp_add_ln<<<T_ALL / 8, 256, 0, stream>>>(
            bx1, bx2, ln1_g + i * 128, ln1_b + i * 128, bh);
        k_gemm128<1,0><<<T_ALL / 64, 256, 0, stream>>>(
            bh, Wf1 + i * 16384, fb1 + i * 128, nullptr, bt, T_ALL, 128);
        k_gemm128<0,1><<<T_ALL / 64, 256, 0, stream>>>(
            bt, Wf2 + i * 16384, fb2 + i * 128, bh, bx1, T_ALL, 128);
        k_ln<<<T_ALL / 8, 256, 0, stream>>>(bx1, ln2_g + i * 128, ln2_b + i * 128, bh);
    }

    k_gemm128<1,0><<<T_ALL / 64, 256, 0, stream>>>(
        bh, Wqw1, qb1, nullptr, bt, T_ALL, 256);
    k_final<<<T_ALL / 8, 256, 0, stream>>>(bt, qw2, qb2, (float*)d_out);
}
